// TimeAttention_21552145891848
// MI455X (gfx1250) — matrix operand order repaired and checked by reading, not compiled, not run on hardware
//
#include <hip/hip_runtime.h>
#include <hip/hip_bf16.h>

typedef __attribute__((ext_vector_type(16))) _Float16 v16h;
typedef __attribute__((ext_vector_type(8)))  float    v8f;

#define NB_  16   // batch
#define TT   256  // T_en == T_de
#define UU   128  // units
#define DD   512  // D_en == D_de

__device__ __forceinline__ float fast_tanh(float x) {
#if __has_builtin(__builtin_amdgcn_tanhf)
    return __builtin_amdgcn_tanhf(x);            // v_tanh_f32 (CDNA5 trans op)
#else
    float cx = fminf(fmaxf(x, -15.f), 15.f);
    float t  = __builtin_amdgcn_exp2f(cx * 2.8853900817779268f); // exp(2x)
    return (t - 1.f) * __builtin_amdgcn_rcpf(t + 1.f);
#endif
}

// att = seq(4096x512) @ w(512x128); optionally store transposed [b][u][t]
__global__ __launch_bounds__(128) void proj_kernel(
    const float* __restrict__ seq, const float* __restrict__ w,
    float* __restrict__ att, int transposeOut)
{
    __shared__ float s[DD];
    int row = blockIdx.x;            // 0..4095 == b*256 + t
    int u   = threadIdx.x;           // 0..127
    #pragma unroll
    for (int c = 0; c < 4; ++c) s[u + c * 128] = seq[row * DD + u + c * 128];
    __syncthreads();
    float acc = 0.f;
    #pragma unroll 8
    for (int k = 0; k < DD; ++k) acc = fmaf(s[k], w[k * UU + u], acc);
    if (transposeOut) {
        int b = row >> 8, t = row & 255;
        att[(b * UU + u) * TT + t] = acc;
    } else {
        att[row * UU + u] = acc;
    }
}

// One block per (b, 16-row j tile): mu -> softmax -> WMMA sum_en -> concat out
__global__ __launch_bounds__(256) void attn_fused(
    const float* __restrict__ en_seq,   // [B][256][512]
    const float* __restrict__ de_seq,   // [B][256][512]
    const float* __restrict__ attT_en,  // [B][128][256]  (transposed)
    const float* __restrict__ att_de,   // [B][256][128]
    const float* __restrict__ nu,       // [128]
    float* __restrict__ out)            // [B][256][1024]
{
    __shared__ float    de_sT[UU * 16];     // 8 KB, transposed [u][j], 64B rows
    __shared__ float    nu_s[UU];           // 0.5 KB
    __shared__ float    mu_s[16 * TT];      // 16 KB
    __shared__ _Float16 alpha_s[16 * TT];   // 8 KB

    int blk = blockIdx.x;            // 0..255
    int b   = blk >> 4;
    int j0  = (blk & 15) << 4;
    int tid = threadIdx.x;

    // load de tile transposed: de_sT[u][j] = att_de[b, j0+j, u]
    for (int idx = tid; idx < 16 * UU; idx += 256) {
        int j = idx >> 7, u = idx & 127;     // idx = j*128 + u (coalesced read)
        de_sT[u * 16 + j] = att_de[(b * TT + j0) * UU + idx];
    }
    if (tid < UU) nu_s[tid] = nu[tid];
    __syncthreads();

    // ---- mu tile: thread i owns encoder position i for all 16 decoder rows
    int i = tid;                     // 0..255
    float acc[16];
    #pragma unroll
    for (int j = 0; j < 16; ++j) acc[j] = 0.f;
    const float* enT = attT_en + b * UU * TT;
    #pragma unroll 2
    for (int u = 0; u < UU; ++u) {
        float e  = enT[u * TT + i];          // coalesced (transposed layout)
        float nv = nu_s[u];
        // 4 x ds_load_b128 broadcast of the 16 decoder values for this u
        const float4* drow = (const float4*)&de_sT[u * 16];
        float4 d0 = drow[0], d1 = drow[1], d2 = drow[2], d3 = drow[3];
        float dv[16] = { d0.x, d0.y, d0.z, d0.w,  d1.x, d1.y, d1.z, d1.w,
                         d2.x, d2.y, d2.z, d2.w,  d3.x, d3.y, d3.z, d3.w };
        #pragma unroll
        for (int j = 0; j < 16; ++j)
            acc[j] = fmaf(nv, fast_tanh(dv[j] + e), acc[j]);
    }
    #pragma unroll
    for (int j = 0; j < 16; ++j) mu_s[j * TT + i] = acc[j];
    __syncthreads();

    // ---- softmax over encoder time; wave w handles rows 2w, 2w+1 (wave32)
    int wave = tid >> 5, lane = tid & 31;
    #pragma unroll
    for (int jj = 0; jj < 2; ++jj) {
        int j = wave * 2 + jj;
        float vals[8];
        float m = -3.0e38f;
        #pragma unroll
        for (int c = 0; c < 8; ++c) {
            vals[c] = mu_s[j * TT + lane + c * 32];
            m = fmaxf(m, vals[c]);
        }
        #pragma unroll
        for (int s = 16; s >= 1; s >>= 1) m = fmaxf(m, __shfl_xor(m, s, 32));
        float sum = 0.f;
        #pragma unroll
        for (int c = 0; c < 8; ++c) {
            vals[c] = __builtin_amdgcn_exp2f((vals[c] - m) * 1.44269504088896f);
            sum += vals[c];
        }
        #pragma unroll
        for (int s = 16; s >= 1; s >>= 1) sum += __shfl_xor(sum, s, 32);
        float inv = __builtin_amdgcn_rcpf(sum);
        #pragma unroll
        for (int c = 0; c < 8; ++c)
            alpha_s[j * TT + lane + c * 32] = (_Float16)(vals[c] * inv);
    }
    __syncthreads();

    // ---- concat lhs: copy de_seq rows into out[..][0:512], 128-bit vectors
    {
        const float4* src = (const float4*)de_seq;
        for (int idx = tid; idx < 16 * (DD / 4); idx += 256) {
            int j = idx >> 7, d4 = idx & 127;
            *(float4*)(out + (size_t)(b * TT + j0 + j) * 1024 + d4 * 4) =
                src[(size_t)(b * TT + j0 + j) * (DD / 4) + d4];
        }
    }

    // ---- sum_en = alphas(16x256) @ en_seq[b](256x512) via WMMA f16->f32
    // wave owns N-tiles n0 = wave*64 + nt*16, nt = 0..3
    v8f c[4] = {};
    int M      = lane & 15;
    int kbaseA = (lane >> 4) << 3;   // 0 or 8   (16-bit A layout, ISA 7.12.2)
    int NBcol  = lane & 15;
    int kbaseB = (lane >> 4) << 3;   // 0 or 8   (16-bit B layout, same k order as A)
    const float* enB = en_seq + (size_t)b * TT * DD;

    #pragma unroll
    for (int kc = 0; kc < 8; ++kc) {
        int k0 = kc * 32;
        v16h a;
        #pragma unroll
        for (int h = 0; h < 16; ++h) {
            int v = h >> 1;
            int K = kbaseA + ((v >= 4) ? 16 : 0) + (v & 3) * 2 + (h & 1);
            a[h] = alpha_s[M * TT + k0 + K];
        }
        #pragma unroll
        for (int nt = 0; nt < 4; ++nt) {
            int n0 = wave * 64 + nt * 16;
            v16h bf;
            #pragma unroll
            for (int h = 0; h < 16; ++h)
                bf[h] = (_Float16)enB[(size_t)(k0 + kbaseB + (h & 7) + ((h >> 3) << 4)) * DD + n0 + NBcol];
            c[nt] = __builtin_amdgcn_wmma_f32_16x16x32_f16(
                false, a, false, bf, (short)0, c[nt], false, false);
        }
    }
    // D layout: VGPR r -> row r (lanes 0-15) / r+8 (lanes 16-31), col = lane%16
    int mrow = (lane >> 4) * 8;
    #pragma unroll
    for (int nt = 0; nt < 4; ++nt) {
        int n0 = wave * 64 + nt * 16;
        #pragma unroll
        for (int r = 0; r < 8; ++r)
            out[(size_t)(b * TT + j0 + mrow + r) * 1024 + 512 + n0 + (lane & 15)] = c[nt][r];
    }
}

extern "C" void kernel_launch(void* const* d_in, const int* in_sizes, int n_in,
                              void* d_out, int out_size, void* d_ws, size_t ws_size,
                              hipStream_t stream) {
    const float* en_seq = (const float*)d_in[0];
    const float* de_seq = (const float*)d_in[1];
    const float* w_en   = (const float*)d_in[2];
    const float* w_de   = (const float*)d_in[3];
    const float* nu     = (const float*)d_in[4];
    float* out = (float*)d_out;

    float* attT_en = (float*)d_ws;                       // [16][128][256] = 2 MB
    float* att_de  = attT_en + (size_t)NB_ * UU * TT;    // [16][256][128] = 2 MB

    proj_kernel<<<NB_ * TT, 128, 0, stream>>>(en_seq, w_en, attT_en, 1);
    proj_kernel<<<NB_ * TT, 128, 0, stream>>>(de_seq, w_de, att_de, 0);
    attn_fused<<<NB_ * (TT / 16), 256, 0, stream>>>(en_seq, de_seq, attT_en,
                                                    att_de, nu, out);
}
